// GatedDeltaNet_50766513438887
// MI455X (gfx1250) — matrix operand order repaired and checked by reading, not compiled, not run on hardware
//
#include <hip/hip_runtime.h>
#include <hip/hip_bf16.h>

// ---------------------------------------------------------------------------
// Gated DeltaNet forward for MI455X (gfx1250, wave32, WMMA + async-to-LDS)
//   B=4, L=2048, HID=2048, NH=6, DK=DV=256, KD=VD=1536, K(conv)=4
// ---------------------------------------------------------------------------

typedef __attribute__((ext_vector_type(16))) __bf16 v16bf;
typedef __attribute__((ext_vector_type(8)))  float  v8f;

union FragU { v16bf v; unsigned u[8]; };

__device__ __forceinline__ unsigned short f2bf(float f) {
    unsigned x = __float_as_uint(f);
    unsigned r = (x + 0x7FFFu + ((x >> 16) & 1u)) >> 16;   // round-to-nearest-even
    return (unsigned short)r;
}

// ---------------------------------------------------------------------------
// f32 -> bf16 bulk conversion (4 elems / thread)
// ---------------------------------------------------------------------------
__global__ __launch_bounds__(256)
void to_bf16_kernel(const float* __restrict__ in, unsigned short* __restrict__ out,
                    int total4)
{
    int i = blockIdx.x * 256 + threadIdx.x;
    if (i >= total4) return;
    float4 f = reinterpret_cast<const float4*>(in)[i];
    union { unsigned short us[4]; uint2 u2; } p;
    p.us[0] = f2bf(f.x); p.us[1] = f2bf(f.y);
    p.us[2] = f2bf(f.z); p.us[3] = f2bf(f.w);
    reinterpret_cast<uint2*>(out)[i] = p.u2;
}

// ---------------------------------------------------------------------------
// W (K x N, f32) -> Wt (N x K, bf16) tiled transpose
// ---------------------------------------------------------------------------
__global__ __launch_bounds__(256)
void transpose_bf16_kernel(const float* __restrict__ W, unsigned short* __restrict__ Wt,
                           int K, int N)
{
    __shared__ unsigned short t[32][33];
    const int tx = threadIdx.x & 31, ty = threadIdx.x >> 5;
    const int n0 = blockIdx.x * 32, k0 = blockIdx.y * 32;
#pragma unroll
    for (int r = 0; r < 32; r += 8)
        t[ty + r][tx] = f2bf(W[(size_t)(k0 + ty + r) * N + n0 + tx]);
    __syncthreads();
#pragma unroll
    for (int r = 0; r < 32; r += 8)
        Wt[(size_t)(n0 + ty + r) * K + k0 + tx] = t[tx][ty + r];
}

// ---------------------------------------------------------------------------
// bf16 WMMA GEMM, double-buffered async global->LDS staging:
//   C(M,N) f32 = A(M,K) bf16  x  Bt(N,K) bf16 (pre-transposed weights)
// Tile 128x128x32, 256 threads / 8 waves; wave computes 32x64 (8 WMMAs/k-step).
// LDS rows stride 40 bf16 (80 B): 16B-aligned for b128, bank-rotated.
// ---------------------------------------------------------------------------
#define TM2 128
#define TN2 128
#define TK2 32
#define LSTR 40

__global__ __launch_bounds__(256)
void gemm_bf16_wmma(const unsigned short* __restrict__ A,
                    const unsigned short* __restrict__ Bt,
                    float* __restrict__ C, int M, int N, int K)
{
    __shared__ __align__(16) unsigned short As[2][TM2 * LSTR]; // 2 x 10240 B
    __shared__ __align__(16) unsigned short Bs[2][TN2 * LSTR]; // 2 x 10240 B

    const int tid  = threadIdx.x;
    const int lane = tid & 31;
    const int wave = tid >> 5;
    const int m0 = blockIdx.y * TM2;
    const int n0 = blockIdx.x * TN2;
    const int wm = (wave & 3) * 32;   // wave's M strip within tile
    const int wn = (wave >> 2) * 64;  // wave's N strip within tile

    v8f acc[2][4] = {};

    // staging: thread covers 16 bf16 (32 B) of one row of each tile
    const int srow = tid >> 1;            // 0..127
    const int skh  = (tid & 1) * 16;      // 0 or 16
    const unsigned aLds[2] = { (unsigned)(size_t)&As[0][srow * LSTR + skh],
                               (unsigned)(size_t)&As[1][srow * LSTR + skh] };
    const unsigned bLds[2] = { (unsigned)(size_t)&Bs[0][srow * LSTR + skh],
                               (unsigned)(size_t)&Bs[1][srow * LSTR + skh] };
    const unsigned short* aG = A  + (size_t)(m0 + srow) * K + skh;
    const unsigned short* bG = Bt + (size_t)(n0 + srow) * K + skh;

    const int rA = lane & 15;
    const int hi = lane >> 4;
    const int T  = K / TK2;

#define GEMM_ISSUE(KT, BUF)                                                     \
    do {                                                                        \
        unsigned long long ga = (unsigned long long)(size_t)(aG + (KT) * TK2);  \
        unsigned long long gb = (unsigned long long)(size_t)(bG + (KT) * TK2);  \
        asm volatile("global_load_async_to_lds_b128 %0, %1, off"                \
                     :: "v"(aLds[BUF]), "v"(ga) : "memory");                    \
        asm volatile("global_load_async_to_lds_b128 %0, %1, off offset:16"      \
                     :: "v"(aLds[BUF]), "v"(ga) : "memory");                    \
        asm volatile("global_load_async_to_lds_b128 %0, %1, off"                \
                     :: "v"(bLds[BUF]), "v"(gb) : "memory");                    \
        asm volatile("global_load_async_to_lds_b128 %0, %1, off offset:16"      \
                     :: "v"(bLds[BUF]), "v"(gb) : "memory");                    \
    } while (0)

    // prologue: stage tile 0
    GEMM_ISSUE(0, 0);

    for (int kt = 0; kt < T; ++kt) {
        const int buf = kt & 1;
        if (kt + 1 < T) {
            GEMM_ISSUE(kt + 1, buf ^ 1);                    // prefetch next tile
            asm volatile("s_wait_asynccnt 0x4" ::: "memory"); // tile kt resident
        } else {
            asm volatile("s_wait_asynccnt 0x0" ::: "memory");
        }
        __syncthreads();   // tile kt visible across all waves

        const unsigned short* Ab = As[buf];
        const unsigned short* Bb = Bs[buf];

        // ---- A fragments (16x32 bf16 ISA layout), 2 M-subtiles
        FragU afr[2];
#pragma unroll
        for (int mi = 0; mi < 2; ++mi) {
            const int row = wm + mi * 16 + rA;
#pragma unroll
            for (int p = 0; p < 8; ++p) {
                int kk = ((p >> 2) << 4) + (hi << 3) + ((p & 3) << 1);
                afr[mi].u[p] = *reinterpret_cast<const unsigned*>(&Ab[row * LSTR + kk]);
            }
        }
        // ---- B fragments (32x16) + 8 WMMAs
#pragma unroll
        for (int nt = 0; nt < 4; ++nt) {
            FragU bfr;
            const int col = wn + nt * 16 + rA;
#pragma unroll
            for (int p = 0; p < 8; ++p) {
                int kk = ((p >> 2) << 4) + (hi << 3) + ((p & 3) << 1);
                bfr.u[p] = *reinterpret_cast<const unsigned*>(&Bb[col * LSTR + kk]);
            }
#pragma unroll
            for (int mi = 0; mi < 2; ++mi)
                acc[mi][nt] = __builtin_amdgcn_wmma_f32_16x16x32_bf16(
                    false, afr[mi].v, false, bfr.v, (short)0, acc[mi][nt], false, false);
        }
        __syncthreads();   // all reads of buf done before it is re-staged
    }
#undef GEMM_ISSUE

    // ---- store C (f32 16x16 layout: row=(lane>>4)*8+p, col=lane&15)
    const int colb = lane & 15;
    const int rowb = (lane >> 4) * 8;
#pragma unroll
    for (int mi = 0; mi < 2; ++mi)
#pragma unroll
        for (int nt = 0; nt < 4; ++nt)
#pragma unroll
            for (int p = 0; p < 8; ++p)
                C[(size_t)(m0 + wm + mi * 16 + rowb + p) * N + n0 + wn + nt * 16 + colb]
                    = acc[mi][nt][p];
}

// ---------------------------------------------------------------------------
// Depthwise causal conv1d (K=4) + SiLU over (B, L, C=1536)
// ---------------------------------------------------------------------------
__global__ __launch_bounds__(256)
void conv_silu_kernel(const float* __restrict__ in, const float* __restrict__ w,
                      float* __restrict__ out, int total)
{
    int idx = blockIdx.x * 256 + threadIdx.x;
    if (idx >= total) return;
    int c = idx % 1536;
    size_t bt = (size_t)idx / 1536;      // b*2048 + t
    int t = (int)(bt & 2047);
    float acc = 0.f;
#pragma unroll
    for (int jj = 0; jj < 4; ++jj) {
        int ts = t + jj - 3;
        float xv = (ts >= 0) ? in[(bt + jj - 3) * 1536 + c] : 0.f;
        acc += w[c * 4 + jj] * xv;
    }
    out[idx] = acc / (1.f + expf(-acc));   // SiLU
}

// ---------------------------------------------------------------------------
// beta = sigmoid(x@Wb), g = -exp(A_log)*softplus(x@Wa + dt_bias)
// one block (256 threads) per (b,t) row; NH=6 outputs each
// ---------------------------------------------------------------------------
__global__ __launch_bounds__(256)
void beta_g_kernel(const float* __restrict__ x, const float* __restrict__ Wb,
                   const float* __restrict__ Wa, const float* __restrict__ A_log,
                   const float* __restrict__ dt_bias,
                   float* __restrict__ beta, float* __restrict__ g)
{
    const int bt  = blockIdx.x;
    const int tid = threadIdx.x;
    const int lane = tid & 31, wave = tid >> 5;
    float accB[6] = {}, accA[6] = {};
    const float* xr = x + (size_t)bt * 2048;
    for (int kk = tid; kk < 2048; kk += 256) {
        float xv = xr[kk];
#pragma unroll
        for (int h = 0; h < 6; ++h) {
            accB[h] += xv * Wb[kk * 6 + h];
            accA[h] += xv * Wa[kk * 6 + h];
        }
    }
#pragma unroll
    for (int h = 0; h < 6; ++h)
        for (int off = 16; off; off >>= 1) {
            accB[h] += __shfl_xor(accB[h], off);
            accA[h] += __shfl_xor(accA[h], off);
        }
    __shared__ float sB[8][6], sA[8][6];
    if (lane == 0)
#pragma unroll
        for (int h = 0; h < 6; ++h) { sB[wave][h] = accB[h]; sA[wave][h] = accA[h]; }
    __syncthreads();
    if (tid < 6) {
        float b = 0.f, a = 0.f;
#pragma unroll
        for (int w = 0; w < 8; ++w) { b += sB[w][tid]; a += sA[w][tid]; }
        beta[(size_t)bt * 6 + tid] = 1.f / (1.f + expf(-b));
        float z = a + dt_bias[tid];
        float sp = (z > 20.f) ? z : log1pf(expf(z));
        g[(size_t)bt * 6 + tid] = -expf(A_log[tid]) * sp;
    }
}

// ---------------------------------------------------------------------------
// Gated delta-rule scan, column-parallel: one wave per state column.
// grid = (DV/8=32, NH=6, B=4), 256 threads (8 waves -> 8 columns / block).
// Each lane holds 8 state floats (rows r = lane + 32*i) in VGPRs.
// k/q rows are ping-pong prefetched with async-to-LDS one step ahead;
// g/beta time series for this (b,h) preloaded to LDS once.
// ---------------------------------------------------------------------------
__global__ __launch_bounds__(256)
void scan_kernel(const float* __restrict__ qb, const float* __restrict__ kb,
                 const float* __restrict__ vb, const float* __restrict__ gb,
                 const float* __restrict__ betab,
                 float* __restrict__ ob, float* __restrict__ fstate)
{
    const int tid = threadIdx.x, lane = tid & 31, wave = tid >> 5;
    const int cb = blockIdx.x;   // column block
    const int h  = blockIdx.y;
    const int b  = blockIdx.z;
    const int j  = cb * 8 + wave;

    __shared__ float lk[2][256], lq[2][256];
    __shared__ float gts[2048], bts[2048];
    __shared__ float red[16];
    __shared__ float scal[4];

    // preload g/beta series for this (b,h)
    for (int t = tid; t < 2048; t += 256) {
        size_t gi = ((size_t)b * 2048 + t) * 6 + h;
        gts[t] = gb[gi];
        bts[t] = betab[gi];
    }

    const unsigned kLds[2] = { (unsigned)(size_t)&lk[0][tid],
                               (unsigned)(size_t)&lk[1][tid] };
    const unsigned qLds[2] = { (unsigned)(size_t)&lq[0][tid],
                               (unsigned)(size_t)&lq[1][tid] };

#define SCAN_ISSUE(T, BUF)                                                        \
    do {                                                                          \
        size_t base_ = ((size_t)b * 2048 + (T)) * 1536 + h * 256 + tid;           \
        unsigned long long ka_ = (unsigned long long)(size_t)(kb + base_);        \
        unsigned long long qa_ = (unsigned long long)(size_t)(qb + base_);        \
        asm volatile("global_load_async_to_lds_b32 %0, %1, off"                   \
                     :: "v"(kLds[BUF]), "v"(ka_) : "memory");                     \
        asm volatile("global_load_async_to_lds_b32 %0, %1, off"                   \
                     :: "v"(qLds[BUF]), "v"(qa_) : "memory");                     \
    } while (0)

    SCAN_ISSUE(0, 0);

    float S[8] = {0.f,0.f,0.f,0.f,0.f,0.f,0.f,0.f};
    const float qscale = 0.0625f;  // DK^-0.5 = 256^-0.5

    for (int t = 0; t < 2048; ++t) {
        const int buf = t & 1;
        const size_t base = ((size_t)b * 2048 + t) * 1536 + h * 256;
        const float vj = vb[base + j];          // broadcast load, overlaps below
        if (t + 1 < 2048) {
            SCAN_ISSUE(t + 1, buf ^ 1);         // prefetch next step's k/q
            asm volatile("s_wait_asynccnt 0x2" ::: "memory"); // step t resident
        } else {
            asm volatile("s_wait_asynccnt 0x0" ::: "memory");
        }
        float kv = lk[buf][tid];                // own slot: own-wave async done
        float qv = lq[buf][tid];
        float sk = kv * kv, sq = qv * qv;
        for (int off = 16; off; off >>= 1) {
            sk += __shfl_xor(sk, off);
            sq += __shfl_xor(sq, off);
        }
        if (lane == 0) { red[wave] = sk; red[8 + wave] = sq; }
        __syncthreads();   // + makes all waves' k/q rows for step t visible
        if (tid == 0) {
            float tk = 0.f, tq = 0.f;
#pragma unroll
            for (int w = 0; w < 8; ++w) { tk += red[w]; tq += red[8 + w]; }
            scal[0] = 1.f    / fmaxf(sqrtf(tk), 1e-12f);
            scal[1] = qscale / fmaxf(sqrtf(tq), 1e-12f);
            scal[2] = expf(gts[t]);
            scal[3] = bts[t];
        }
        __syncthreads();
        const float rk = scal[0], rq = scal[1], eg = scal[2], bet = scal[3];

        float kn[8]; float c = 0.f;
#pragma unroll
        for (int i = 0; i < 8; ++i) { kn[i] = lk[buf][lane + i * 32] * rk; c += kn[i] * S[i]; }
        for (int off = 16; off; off >>= 1) c += __shfl_xor(c, off);
        const float u = bet * (vj - eg * c);

        float o = 0.f;
#pragma unroll
        for (int i = 0; i < 8; ++i) {
            S[i] = eg * S[i] + kn[i] * u;
            o += (lq[buf][lane + i * 32] * rq) * S[i];
        }
        for (int off = 16; off; off >>= 1) o += __shfl_xor(o, off);
        if (lane == 0) ob[base + j] = o;
        __syncthreads();   // reads of buf done before step t+2 re-stages it
    }
#undef SCAN_ISSUE

    // final_state (B, NH, DK, DV)
#pragma unroll
    for (int i = 0; i < 8; ++i) {
        int r = lane + i * 32;
        fstate[(((size_t)b * 6 + h) * 256 + r) * 256 + j] = S[i];
    }
}

// ---------------------------------------------------------------------------
// Gated RMSNorm, fused bf16 output for the final WMMA GEMM:
//   out_bf = bf16( o * rsqrt(mean(o^2)+eps) * w * silu(gate) )
// one block per (b,t,h), 256 threads (DV=256)
// ---------------------------------------------------------------------------
__global__ __launch_bounds__(256)
void gated_rmsnorm_kernel(const float* __restrict__ o, const float* __restrict__ gpre,
                          const float* __restrict__ nw, unsigned short* __restrict__ obf)
{
    const int tid = threadIdx.x, lane = tid & 31, wave = tid >> 5;
    const int bt = blockIdx.x / 6, h = blockIdx.x % 6;
    const size_t idx = (size_t)bt * 1536 + h * 256 + tid;
    float v = o[idx];
    float ss = v * v;
    for (int off = 16; off; off >>= 1) ss += __shfl_xor(ss, off);
    __shared__ float red[8];
    if (lane == 0) red[wave] = ss;
    __syncthreads();
    if (tid == 0) {
        float s = 0.f;
#pragma unroll
        for (int w = 0; w < 8; ++w) s += red[w];
        red[0] = s;
    }
    __syncthreads();
    const float r = rsqrtf(red[0] * (1.f / 256.f) + 1e-5f);
    const float gt = gpre[idx];
    obf[idx] = f2bf(v * r * nw[tid] * (gt / (1.f + expf(-gt))));
}

// ---------------------------------------------------------------------------
extern "C" void kernel_launch(void* const* d_in, const int* in_sizes, int n_in,
                              void* d_out, int out_size, void* d_ws, size_t ws_size,
                              hipStream_t stream)
{
    const float* x       = (const float*)d_in[0];
    const float* Wq      = (const float*)d_in[1];
    const float* Wk      = (const float*)d_in[2];
    const float* Wv      = (const float*)d_in[3];
    const float* Wb      = (const float*)d_in[4];
    const float* Wa      = (const float*)d_in[5];
    const float* A_log   = (const float*)d_in[6];
    const float* dt_bias = (const float*)d_in[7];
    const float* conv_q  = (const float*)d_in[8];
    const float* conv_k  = (const float*)d_in[9];
    const float* conv_v  = (const float*)d_in[10];
    const float* Wg      = (const float*)d_in[11];
    const float* norm_w  = (const float*)d_in[12];
    const float* Wo      = (const float*)d_in[13];

    float* out    = (float*)d_out;
    float* fstate = out + (size_t)4 * 2048 * 2048;   // final_state tail

    float* ws = (float*)d_ws;
    const size_t NB  = (size_t)4 * 2048 * 1536;      // 12,582,912 floats
    const size_t NX  = (size_t)8192 * 2048;          // x elems
    const size_t NW  = (size_t)2048 * 1536;          // per-weight elems
    float* qpre = ws;
    float* kpre = ws + NB;
    float* vpre = ws + 2 * NB;
    float* qc   = ws + 3 * NB;
    float* kc   = ws + 4 * NB;
    float* vc   = ws + 5 * NB;
    float* beta = ws + 6 * NB;                       // B*L*NH = 49,152
    float* gdec = beta + (size_t)4 * 2048 * 6;
    unsigned short* x_bf = (unsigned short*)(gdec + (size_t)4 * 2048 * 6);
    unsigned short* Wqt  = x_bf + NX;
    unsigned short* Wkt  = Wqt + NW;
    unsigned short* Wvt  = Wkt + NW;
    unsigned short* Wgt  = Wvt + NW;
    unsigned short* Wot  = Wgt + NW;
    unsigned short* on_bf = (unsigned short*)vpre;   // reuse after conv_v (24MB<=48MB)
    float* gpre = kpre;                              // reuse after conv_k
    float* obuf = qpre;                              // reuse after conv_q

    const dim3 blk(256);

    // 0: one-shot bf16 conversions / weight transposes
    to_bf16_kernel<<<(int)((NX / 4 + 255) / 256), blk, 0, stream>>>(x, x_bf, (int)(NX / 4));
    const dim3 gT1(1536 / 32, 2048 / 32);            // (N/32, K/32) for HIDxKD weights
    transpose_bf16_kernel<<<gT1, blk, 0, stream>>>(Wq, Wqt, 2048, 1536);
    transpose_bf16_kernel<<<gT1, blk, 0, stream>>>(Wk, Wkt, 2048, 1536);
    transpose_bf16_kernel<<<gT1, blk, 0, stream>>>(Wv, Wvt, 2048, 1536);
    transpose_bf16_kernel<<<gT1, blk, 0, stream>>>(Wg, Wgt, 2048, 1536);
    const dim3 gT2(2048 / 32, 1536 / 32);
    transpose_bf16_kernel<<<gT2, blk, 0, stream>>>(Wo, Wot, 1536, 2048);

    // 1-3: input projections (bf16 WMMA, double-buffered async staging)
    const dim3 gProj(1536 / TN2, 8192 / TM2);
    gemm_bf16_wmma<<<gProj, blk, 0, stream>>>(x_bf, Wqt, qpre, 8192, 1536, 2048);
    gemm_bf16_wmma<<<gProj, blk, 0, stream>>>(x_bf, Wkt, kpre, 8192, 1536, 2048);
    gemm_bf16_wmma<<<gProj, blk, 0, stream>>>(x_bf, Wvt, vpre, 8192, 1536, 2048);

    // 4-6: causal depthwise conv + SiLU
    const int totc = 4 * 2048 * 1536;
    const int gc = (totc + 255) / 256;
    conv_silu_kernel<<<gc, blk, 0, stream>>>(qpre, conv_q, qc, totc);
    conv_silu_kernel<<<gc, blk, 0, stream>>>(kpre, conv_k, kc, totc);
    conv_silu_kernel<<<gc, blk, 0, stream>>>(vpre, conv_v, vc, totc);

    // 7: gate projection (into reused kpre)
    gemm_bf16_wmma<<<gProj, blk, 0, stream>>>(x_bf, Wgt, gpre, 8192, 1536, 2048);

    // 8: beta / decay
    beta_g_kernel<<<8192, blk, 0, stream>>>(x, Wb, Wa, A_log, dt_bias, beta, gdec);

    // 9: column-parallel gated delta-rule scan (obuf reuses qpre)
    scan_kernel<<<dim3(32, 6, 4), blk, 0, stream>>>(qc, kc, vc, gdec, beta, obuf, fstate);

    // 10: gated RMSNorm, fused bf16 conversion (on_bf reuses vpre)
    gated_rmsnorm_kernel<<<8192 * 6, blk, 0, stream>>>(obuf, gpre, norm_w, on_bf);

    // 11: output projection
    const dim3 gOut(2048 / TN2, 8192 / TM2);
    gemm_bf16_wmma<<<gOut, blk, 0, stream>>>(on_bf, Wot, out, 8192, 2048, 1536);
}
